// ScaledDotProductAttention_5119601016845
// MI455X (gfx1250) — matrix operand order repaired and checked by reading, not compiled, not run on hardware
//
#include <hip/hip_runtime.h>
#include <math.h>

// ---------------- types ----------------
typedef __attribute__((ext_vector_type(16))) __bf16 v16bf;
typedef __attribute__((ext_vector_type(8)))  float  v8f;
typedef __attribute__((ext_vector_type(4)))  float  f4;
typedef __attribute__((ext_vector_type(4)))  unsigned int u32x4;

// ---------------- problem constants (match reference) ----------------
#define NB   32      // batch
#define SQ   2048    // query length
#define SK   2048    // key length
#define DH   64      // head dim
#define SSTRIDE 2064 // LDS score-strip row stride in floats (16-float pad: bank-conflict free)
#define VSTRIDE 40   // transposed V stage row stride in bf16 (multiple of 8 -> 16B aligned rows)
#define VSLOT   (DH * VSTRIDE)  // bf16 elems per wave's V slot
#define OST     68   // obuf row stride in floats
#define NEG_BIG (-1e10f)

#define SMEM_BYTES (16*SSTRIDE*4 /*strip*/ + 4*VSLOT*2 /*vstage/obuf*/ + 128*4 + 128*4 /*reductions*/)

__device__ __forceinline__ v16bf pack16(f4 f0, f4 f1, f4 f2, f4 f3) {
  v16bf a;
  a[0]=(__bf16)f0.x;  a[1]=(__bf16)f0.y;  a[2]=(__bf16)f0.z;  a[3]=(__bf16)f0.w;
  a[4]=(__bf16)f1.x;  a[5]=(__bf16)f1.y;  a[6]=(__bf16)f1.z;  a[7]=(__bf16)f1.w;
  a[8]=(__bf16)f2.x;  a[9]=(__bf16)f2.y;  a[10]=(__bf16)f2.z; a[11]=(__bf16)f2.w;
  a[12]=(__bf16)f3.x; a[13]=(__bf16)f3.y; a[14]=(__bf16)f3.z; a[15]=(__bf16)f3.w;
  return a;
}

__device__ __forceinline__ v16bf load_v16bf_lds(const __bf16* p) {
  union { f4 q[2]; v16bf v; } u;
  u.q[0] = *(const f4*)(p);
  u.q[1] = *(const f4*)(p + 8);
  return u.v;
}

__device__ __forceinline__ f4 apply_mask(f4 x, unsigned mw) {
  if (mw & 0x000000FFu) x.x = NEG_BIG;
  if (mw & 0x0000FF00u) x.y = NEG_BIG;
  if (mw & 0x00FF0000u) x.z = NEG_BIG;
  if (mw & 0xFF000000u) x.w = NEG_BIG;
  return x;
}

// CDNA5 async DMA: push 16B per lane from LDS to global, tracked by ASYNCcnt.
__device__ __forceinline__ void async_store_b128(const float* gaddr, unsigned lds_off) {
  asm volatile("global_store_async_from_lds_b128 %0, %1, off"
               :: "v"((unsigned long long)(uintptr_t)gaddr), "v"(lds_off)
               : "memory");
}

__global__ __launch_bounds__(128) void
attn_fwd_kernel(const float* __restrict__ Q, const float* __restrict__ K,
                const float* __restrict__ V, const unsigned char* __restrict__ M,
                float* __restrict__ Out, float* __restrict__ Attn, float scale)
{
  extern __shared__ char smem[];
  float*  strip  = (float*)smem;                                   // [16][SSTRIDE] f32 scores/probs
  __bf16* vstage = (__bf16*)(smem + 16*SSTRIDE*4);                 // [4 waves][DH][VSTRIDE] bf16 V^T
  float*  obuf   = (float*)vstage;                                 // overlay: [4][16][OST] partial O
  float*  redmax = (float*)(smem + 16*SSTRIDE*4 + 4*VSLOT*2);      // [128]
  float*  redsum = redmax + 128;                                   // [128]

  const int t    = threadIdx.x;
  const int lane = t & 31;
  const int w    = t >> 5;
  const int b    = blockIdx.y;
  const int qbase = blockIdx.x * 16;
  const int nlo  = lane & 15;
  const bool hi  = lane >= 16;
  const int m0   = hi ? 8 : 0;

  // ---------------- load Q tile as A-operands (16x32 bf16 layout, two K-chunks) ----------------
  const float* qrow = Q + ((size_t)b * SQ + qbase + nlo) * DH;
  v16bf a0, a1;
  {
    const int koff = hi ? 8 : 0;
    a0 = pack16(*(const f4*)(qrow + koff),      *(const f4*)(qrow + koff + 4),
                *(const f4*)(qrow + koff + 16), *(const f4*)(qrow + koff + 20));
    a1 = pack16(*(const f4*)(qrow + 32 + koff),      *(const f4*)(qrow + 32 + koff + 4),
                *(const f4*)(qrow + 32 + koff + 16), *(const f4*)(qrow + 32 + koff + 20));
  }

  // ---------------- phase 1: S = scale * Q K^T into LDS strip (mask applied later) ----------------
  for (int kt = w; kt < SK/16; kt += 4) {
    const float* krow = K + ((size_t)b * SK + kt*16 + nlo) * DH;
    if (kt + 4 < SK/16)
      __builtin_prefetch(K + ((size_t)b * SK + (kt+4)*16 + nlo) * DH, 0, 3);
    const int db = hi ? 8 : 0;
    v16bf b0 = pack16(*(const f4*)(krow + db),      *(const f4*)(krow + db + 4),
                      *(const f4*)(krow + db + 16),  *(const f4*)(krow + db + 20));
    v16bf b1 = pack16(*(const f4*)(krow + 32 + db),     *(const f4*)(krow + 32 + db + 4),
                      *(const f4*)(krow + 32 + db + 16), *(const f4*)(krow + 32 + db + 20));
    v8f acc = {0.f,0.f,0.f,0.f,0.f,0.f,0.f,0.f};
    acc = __builtin_amdgcn_wmma_f32_16x16x32_bf16(false, a0, false, b0, (short)0, acc, false, false);
    acc = __builtin_amdgcn_wmma_f32_16x16x32_bf16(false, a1, false, b1, (short)0, acc, false, false);

    float* sp = strip + m0*SSTRIDE + kt*16 + nlo;
    #pragma unroll
    for (int r = 0; r < 8; ++r)
      sp[r * SSTRIDE] = acc[r] * scale;
  }
  __syncthreads();

  // ---------------- phase 2: masked softmax over the strip ----------------
  const int row = t & 15;       // 8 threads per row
  const int seg = t >> 4;       // each owns 256 columns
  float* srow = strip + row*SSTRIDE + seg*256;
  const unsigned char* mrow = M + ((size_t)b * SQ + qbase + row) * SK + seg*256;

  // max pass: coalesced 16B mask loads, apply mask, write masked score back
  float mx = -3.4e38f;
  #pragma unroll 2
  for (int i = 0; i < 256; i += 16) {
    u32x4 mb = *(const u32x4*)(mrow + i);
    f4 x0 = apply_mask(*(const f4*)(srow + i),      mb.x);
    f4 x1 = apply_mask(*(const f4*)(srow + i + 4),  mb.y);
    f4 x2 = apply_mask(*(const f4*)(srow + i + 8),  mb.z);
    f4 x3 = apply_mask(*(const f4*)(srow + i + 12), mb.w);
    *(f4*)(srow + i)      = x0;
    *(f4*)(srow + i + 4)  = x1;
    *(f4*)(srow + i + 8)  = x2;
    *(f4*)(srow + i + 12) = x3;
    f4 m01 = {fmaxf(x0.x,x1.x), fmaxf(x0.y,x1.y), fmaxf(x0.z,x1.z), fmaxf(x0.w,x1.w)};
    f4 m23 = {fmaxf(x2.x,x3.x), fmaxf(x2.y,x3.y), fmaxf(x2.z,x3.z), fmaxf(x2.w,x3.w)};
    mx = fmaxf(mx, fmaxf(fmaxf(fmaxf(m01.x,m01.y), fmaxf(m01.z,m01.w)),
                         fmaxf(fmaxf(m23.x,m23.y), fmaxf(m23.z,m23.w))));
  }
  redmax[t] = mx;
  __syncthreads();
  float rmx = redmax[row];
  #pragma unroll
  for (int s2 = 1; s2 < 8; ++s2) rmx = fmaxf(rmx, redmax[row + 16*s2]);

  // exp pass
  float sum = 0.f;
  #pragma unroll 2
  for (int i = 0; i < 256; i += 4) {
    f4 x = *(const f4*)(srow + i);
    x.x = __expf(x.x - rmx); x.y = __expf(x.y - rmx);
    x.z = __expf(x.z - rmx); x.w = __expf(x.w - rmx);
    sum += (x.x + x.y) + (x.z + x.w);
    *(f4*)(srow + i) = x;
  }
  redsum[t] = sum;
  __syncthreads();
  float rs = 0.f;
  #pragma unroll
  for (int s2 = 0; s2 < 8; ++s2) rs += redsum[row + 16*s2];
  const float inv = 1.0f / rs;

  // ---------------- phase 3: normalize in place, then async DMA LDS -> Attn ----------------
  #pragma unroll 4
  for (int i = 0; i < 256; i += 4) {
    f4 x = *(const f4*)(srow + i);
    x.x *= inv; x.y *= inv; x.z *= inv; x.w *= inv;
    *(f4*)(srow + i) = x;
  }
  // ensure in-place LDS stores are visible to the async engine before it reads
  asm volatile("s_wait_dscnt 0" ::: "memory");
  {
    const float* arow = Attn + ((size_t)b * SQ + qbase + row) * SK + seg*256;
    const unsigned lds_off = (unsigned)(uintptr_t)(const void*)srow;
    #pragma unroll 4
    for (int i = 0; i < 256; i += 4)
      async_store_b128(arow + i, lds_off + (unsigned)(i * 4));
  }
  __syncthreads();   // strip now fully normalized for every wave's P@V reads

  // ---------------- phase 4: O = P_norm V ----------------
  v8f oacc[4];
  #pragma unroll
  for (int dt = 0; dt < 4; ++dt) oacc[dt] = (v8f){0.f,0.f,0.f,0.f,0.f,0.f,0.f,0.f};

  __bf16* vt = vstage + w * VSLOT;    // this wave's private V^T staging slot
  const int kb2 = hi ? 16 : 0;

  for (int c2 = w; c2 < SK/32; c2 += 4) {
    const int nb0 = c2 * 32;
    // stage 32x64 f32 V chunk into LDS transposed as bf16: vt[d][n]
    #pragma unroll 4
    for (int it = 0; it < 16; ++it) {
      int idx = it*32 + lane;
      int vr = idx >> 4, vc4 = idx & 15;
      f4 f = *(const f4*)(V + ((size_t)b * SK + nb0 + vr) * DH + vc4*4);
      vt[(vc4*4 + 0)*VSTRIDE + vr] = (__bf16)f.x;
      vt[(vc4*4 + 1)*VSTRIDE + vr] = (__bf16)f.y;
      vt[(vc4*4 + 2)*VSTRIDE + vr] = (__bf16)f.z;
      vt[(vc4*4 + 3)*VSTRIDE + vr] = (__bf16)f.w;
    }
    // A-operand: 16x32 normalized prob chunk from the strip
    const int nb = nb0 + (hi ? 16 : 0);
    const float* pr = strip + nlo*SSTRIDE + nb;
    v16bf pa = pack16(*(const f4*)(pr),      *(const f4*)(pr + 4),
                      *(const f4*)(pr + 8), *(const f4*)(pr + 12));
    #pragma unroll
    for (int dt = 0; dt < 4; ++dt) {
      v16bf vb = load_v16bf_lds(vt + (dt*16 + nlo)*VSTRIDE + kb2);
      oacc[dt] = __builtin_amdgcn_wmma_f32_16x16x32_bf16(false, pa, false, vb, (short)0, oacc[dt], false, false);
    }
  }

  __syncthreads();                    // all waves done with vstage -> safe to overlay obuf

  // ---------------- phase 5: cross-wave O reduction + final write ----------------
  {
    float* ob = obuf + w * 16 * OST;
    #pragma unroll
    for (int dt = 0; dt < 4; ++dt)
      #pragma unroll
      for (int r = 0; r < 8; ++r)
        ob[(r + m0)*OST + dt*16 + nlo] = oacc[dt][r];
  }
  __syncthreads();

  #pragma unroll
  for (int i = 0; i < 8; ++i) {
    int fidx = t + 128*i;             // 16*64 = 1024 outputs / 128 threads
    int orow = fidx >> 6, ocol = fidx & 63;
    float o = obuf[orow*OST + ocol] + obuf[(16 + orow)*OST + ocol]
            + obuf[(32 + orow)*OST + ocol] + obuf[(48 + orow)*OST + ocol];
    Out[((size_t)b * SQ + qbase + orow) * DH + ocol] = o;
  }

  // drain async LDS->global DMA (s_endpgm also implies wait-idle)
  asm volatile("s_wait_asynccnt 0" ::: "memory");
}

extern "C" void kernel_launch(void* const* d_in, const int* in_sizes, int n_in,
                              void* d_out, int out_size, void* d_ws, size_t ws_size,
                              hipStream_t stream) {
  (void)in_sizes; (void)n_in; (void)out_size; (void)d_ws; (void)ws_size;
  const float* Q = (const float*)d_in[0];
  const float* K = (const float*)d_in[1];
  const float* V = (const float*)d_in[2];
  const unsigned char* M = (const unsigned char*)d_in[3];
  float* Out  = (float*)d_out;                       // [B,SQ,DH] first
  float* Attn = Out + (size_t)NB * SQ * DH;          // then [B,SQ,SK]

  // scale = (1/sqrt(d)) * log_32(Sk + 1)
  const float scale = (1.0f / 8.0f) * (logf((float)(SK + 1)) / logf(32.0f));

  hipFuncSetAttribute((const void*)attn_fwd_kernel,
                      hipFuncAttributeMaxDynamicSharedMemorySize, SMEM_BYTES);

  dim3 grid(SQ / 16, NB);
  attn_fwd_kernel<<<grid, 128, SMEM_BYTES, stream>>>(Q, K, V, M, Out, Attn, scale);
}
